// selfAttention_71176198029558
// MI455X (gfx1250) — matrix operand order repaired and checked by reading, not compiled, not run on hardware
//
#include <hip/hip_runtime.h>
#include <hip/hip_bf16.h>

typedef __attribute__((ext_vector_type(16))) _Float16 v16h;
typedef __attribute__((ext_vector_type(8)))  float    v8f;
typedef __attribute__((ext_vector_type(4)))  unsigned int v4u;
typedef __attribute__((ext_vector_type(8)))  int      v8i;
typedef __attribute__((ext_vector_type(4)))  int      v4i;

#define NB     2
#define SEQ    2048
#define EMB    1024
#define HEADS  16
#define HDIM   64

// ---------------------------------------------------------------------------
// WMMA fragment helpers (CDNA5 16x16x32 f16 layouts, wave32)
// A (16x32): lane -> row M = lane%16; half=lane/16;
//   vgpr j in 0..3 : K = 8*half + 2j,+1 ; j in 4..7 : K = 16 + 8*half + 2(j-4),+1
// B (32x16): lane -> col N = lane%16; vgpr j: same K mapping as A
// C (16x16): vgpr r -> row M = r + 8*(lane/16), col N = lane%16
// ---------------------------------------------------------------------------

__device__ __forceinline__ v16h load_a_frag(const _Float16* base, int pitch) {
    const int lane = threadIdx.x & 31;
    const int half = lane >> 4, ln = lane & 15;
    const _Float16* r = base + ln * pitch;
    v16h a;
#pragma unroll
    for (int j = 0; j < 4; ++j) {
        int kk = 8 * half + 2 * j;
        a[2 * j]     = r[kk];
        a[2 * j + 1] = r[kk + 1];
    }
#pragma unroll
    for (int j = 4; j < 8; ++j) {
        int kk = 16 + 8 * half + 2 * (j - 4);
        a[2 * j]     = r[kk];
        a[2 * j + 1] = r[kk + 1];
    }
    return a;
}

__device__ __forceinline__ v16h load_b_frag(const _Float16* base, int pitch, int ncol_off) {
    const int lane = threadIdx.x & 31;
    const int half = lane >> 4, ln = lane & 15;
    const _Float16* r = base + (ncol_off + ln) * pitch;
    v16h b;
#pragma unroll
    for (int j = 0; j < 8; ++j) {
        int kk = (j < 4) ? (8 * half + 2 * j) : (16 + 8 * half + 2 * (j - 4));
        b[2 * j]     = r[kk];
        b[2 * j + 1] = r[kk + 1];
    }
    return b;
}

__device__ __forceinline__ v8f wmma16x16x32(v16h a, v16h b, v8f c) {
    return __builtin_amdgcn_wmma_f32_16x16x32_f16(false, a, false, b, (short)0, c, false, false);
}

// ---------------------------------------------------------------------------
// Pre-pass 1: pack mask (int32 0/1) to bit-mask, one u32 per 32 keys (wave ballot)
// ---------------------------------------------------------------------------
__global__ __launch_bounds__(256) void packmask_kernel(const int* __restrict__ mask,
                                                       unsigned* __restrict__ pm) {
    int gid  = blockIdx.x * 256 + threadIdx.x;
    int lane = gid & 31;
    unsigned long long b = __ballot(mask[gid] != 0);
    if (lane == 0) pm[gid >> 5] = (unsigned)b;
}

// ---------------------------------------------------------------------------
// Pre-pass 2: f32 -> f16 conversions (W_fc; and k,v for the TDM/WMMA path)
// ---------------------------------------------------------------------------
__global__ __launch_bounds__(256) void convw_kernel(const float* __restrict__ W,
                                                    _Float16* __restrict__ Wh) {
    int i = (blockIdx.x * 256 + threadIdx.x) * 4;
    float4 v = *(const float4*)(W + i);
    Wh[i + 0] = (_Float16)v.x;  Wh[i + 1] = (_Float16)v.y;
    Wh[i + 2] = (_Float16)v.z;  Wh[i + 3] = (_Float16)v.w;
}

__global__ __launch_bounds__(256) void convkv_kernel(const float* __restrict__ k,
                                                     const float* __restrict__ v,
                                                     _Float16* __restrict__ Kh,
                                                     _Float16* __restrict__ Vh) {
    int i = (blockIdx.x * 256 + threadIdx.x) * 4;
    float4 a = *(const float4*)(k + i);
    Kh[i + 0] = (_Float16)a.x;  Kh[i + 1] = (_Float16)a.y;
    Kh[i + 2] = (_Float16)a.z;  Kh[i + 3] = (_Float16)a.w;
    float4 b = *(const float4*)(v + i);
    Vh[i + 0] = (_Float16)b.x;  Vh[i + 1] = (_Float16)b.y;
    Vh[i + 2] = (_Float16)b.z;  Vh[i + 3] = (_Float16)b.w;
}

// ---------------------------------------------------------------------------
// Kernel: fused flash attention. Block = one (n,h) x 128 query rows, 8 waves.
// K tile staged by the Tensor Data Mover (with LDS row padding done by TDM),
// V tile staged manually with transpose. Writes f16 X[n*SEQ+q][h*64+d].
// ---------------------------------------------------------------------------
__global__ __launch_bounds__(256) void attn_kernel(const float* __restrict__ q,
                                                   const _Float16* __restrict__ Kh,
                                                   const _Float16* __restrict__ Vh,
                                                   const unsigned* __restrict__ pm,
                                                   _Float16* __restrict__ Xout) {
    __shared__ _Float16 Ks[32][72];      // [key][d]   72-half pitch (36 dwords)
    __shared__ _Float16 Vt[64][40];      // [d][key]   20-dword pitch
    __shared__ _Float16 Pb[8][16][40];   // per-wave P tile [qrow][keycol]

    const int t    = threadIdx.x;
    const int lane = t & 31, wv = t >> 5;
    const int half = lane >> 4, ln = lane & 15;

    int blk = blockIdx.x;
    const int qblk = blk & 15;  blk >>= 4;   // SEQ/128 = 16
    const int h    = blk & 15;  blk >>= 4;   // HEADS
    const int n    = blk;
    const int q0   = qblk * 128;

    const size_t headoff = (size_t)n * SEQ * EMB + (size_t)h * HDIM;

    // ---- Q fragments straight from global f32 (one-time per block) ----
    v16h aq0, aq1;
    {
        const float* qr = q + headoff + (size_t)(q0 + 16 * wv + ln) * EMB;
#pragma unroll
        for (int j = 0; j < 4; ++j) {
            int kk = 8 * half + 2 * j;
            aq0[2 * j] = (_Float16)qr[kk];      aq0[2 * j + 1] = (_Float16)qr[kk + 1];
            aq1[2 * j] = (_Float16)qr[kk + 32]; aq1[2 * j + 1] = (_Float16)qr[kk + 33];
        }
#pragma unroll
        for (int j = 4; j < 8; ++j) {
            int kk = 16 + 8 * half + 2 * (j - 4);
            aq0[2 * j] = (_Float16)qr[kk];      aq0[2 * j + 1] = (_Float16)qr[kk + 1];
            aq1[2 * j] = (_Float16)qr[kk + 32]; aq1[2 * j + 1] = (_Float16)qr[kk + 33];
        }
    }

    v8f o0 = {}, o1 = {}, o2 = {}, o3 = {};
    float mrow[8], lrow[8];
#pragma unroll
    for (int r = 0; r < 8; ++r) { mrow[r] = -1e30f; lrow[r] = 0.0f; }

    const float invs = 0.03125f;  // 1/sqrt(1024)

    // TDM descriptor invariants for the K tile (16-bit data, 64 x 32 tile,
    // tensor row stride EMB elements, LDS pad 4 dwords every 32 dwords -> 72-half pitch)
    const unsigned ldsK = (unsigned)(size_t)&Ks[0][0];
    const unsigned long long gaBase =
        (unsigned long long)(size_t)(Kh + headoff);

    for (int kt = 0; kt < SEQ / 32; ++kt) {
        __syncthreads();

        if (wv == 0) {
            unsigned long long ga = gaBase + (unsigned long long)kt * 32ull * EMB * 2ull;
            v4u g0;
            g0[0] = 1u;                                   // count=1, user descriptor
            g0[1] = ldsK;                                 // lds_addr
            g0[2] = (unsigned)(ga & 0xffffffffull);       // global_addr[31:0]
            g0[3] = (unsigned)((ga >> 32) & 0x1ffffffull) // global_addr[56:32]
                    | (2u << 30);                         // type=2 (image)
            v8i g1;
            g1[0] = (int)((1u << 16)      // data_size = 2 bytes
                        | (1u << 20)      // pad_enable
                        | (4u << 22)      // pad_interval: 32 dwords
                        | (3u << 25));    // pad_amount: 4 dwords
            g1[1] = (int)(64u << 16);     // tensor_dim0 = 64 (bits 79:48, low half)
            g1[2] = (int)(32u << 16);     // tensor_dim1 = 32 (bits 111:80, low half)
            g1[3] = (int)(64u << 16);     // tile_dim0 = 64   (bits 127:112)
            g1[4] = 32;                   // tile_dim1 = 32, tile_dim2 = 0
            g1[5] = 1024;                 // tensor_dim0_stride = EMB elements
            g1[6] = 0;                    // stride0 hi / stride1 lo
            g1[7] = 0;
            v4i gz  = {};
            v8i gz8 = {};
            __builtin_amdgcn_tensor_load_to_lds(g0, g1, gz, gz, gz8, 0);
        }

        // ---- stage transposed V tile from f16 copy (1 x b128 per thread) ----
        {
            int i = t;                       // 32 rows x 8 chunks = 256 work items
            int row = i >> 3, dc = (i & 7) * 8;
            const _Float16* src = Vh + headoff + (size_t)(kt * 32 + row) * EMB + dc;
#pragma unroll
            for (int j = 0; j < 8; ++j) Vt[dc + j][row] = src[j];
        }

        if (wv == 0) __builtin_amdgcn_s_wait_tensorcnt(0);
        __syncthreads();

        // ---- S = Q * K^T : 16 queries x 32 keys, contraction over d=64 ----
        v8f z = {};
        v16h b;
        b = load_b_frag(&Ks[0][0], 72, 0);
        v8f s0 = wmma16x16x32(aq0, b, z);
        b = load_b_frag(&Ks[0][32], 72, 0);
        s0 = wmma16x16x32(aq1, b, s0);
        b = load_b_frag(&Ks[0][0], 72, 16);
        v8f s1 = wmma16x16x32(aq0, b, z);
        b = load_b_frag(&Ks[0][32], 72, 16);
        s1 = wmma16x16x32(aq1, b, s1);

        // ---- packed mask + scale + online softmax ----
#pragma unroll
        for (int r = 0; r < 8; ++r) {
            const int qg = q0 + 16 * wv + r + 8 * half;
            const unsigned mbits = pm[((size_t)n * SEQ + qg) * (SEQ / 32) + kt];
            float e0 = ((mbits >> ln) & 1u)        ? s0[r] * invs : -3.0e18f;
            float e1 = ((mbits >> (ln + 16)) & 1u) ? s1[r] * invs : -3.0e18f;

            float tm = fmaxf(e0, e1);
#pragma unroll
            for (int off = 8; off > 0; off >>= 1)
                tm = fmaxf(tm, __shfl_xor(tm, off, 16));
            const float mnew  = fmaxf(mrow[r], tm);
            const float alpha = __expf(mrow[r] - mnew);
            const float p0 = __expf(e0 - mnew);
            const float p1 = __expf(e1 - mnew);
            float ts = p0 + p1;
#pragma unroll
            for (int off = 8; off > 0; off >>= 1)
                ts += __shfl_xor(ts, off, 16);
            lrow[r] = lrow[r] * alpha + ts;
            mrow[r] = mnew;
            o0[r] *= alpha; o1[r] *= alpha; o2[r] *= alpha; o3[r] *= alpha;

            Pb[wv][r + 8 * half][ln]      = (_Float16)p0;
            Pb[wv][r + 8 * half][ln + 16] = (_Float16)p1;
        }
        // within-wave LDS transpose of P: stores must land before gathers
        asm volatile("s_wait_dscnt 0" ::: "memory");

        // ---- O += P * V : contraction over 32 keys, 64 output dims ----
        v16h ap = load_a_frag(&Pb[wv][0][0], 40);
        b = load_b_frag(&Vt[0][0],  40, 0);  o0 = wmma16x16x32(ap, b, o0);
        b = load_b_frag(&Vt[16][0], 40, 0);  o1 = wmma16x16x32(ap, b, o1);
        b = load_b_frag(&Vt[32][0], 40, 0);  o2 = wmma16x16x32(ap, b, o2);
        b = load_b_frag(&Vt[48][0], 40, 0);  o3 = wmma16x16x32(ap, b, o3);
    }

    // ---- normalize, emit f16 rows of X ([n,q,h,d] == row-major [4096][1024]) ----
#pragma unroll
    for (int r = 0; r < 8; ++r) {
        const float il = 1.0f / lrow[r];
        const int qg = q0 + 16 * wv + r + 8 * half;
        _Float16* xp = Xout + ((size_t)n * SEQ + qg) * EMB + h * HDIM;
        xp[ln]      = (_Float16)(o0[r] * il);
        xp[16 + ln] = (_Float16)(o1[r] * il);
        xp[32 + ln] = (_Float16)(o2[r] * il);
        xp[48 + ln] = (_Float16)(o3[r] * il);
    }
}

// ---------------------------------------------------------------------------
// FC: out = X @ W^T + b. Block = 8 waves, tile 128 rows x 64 cols.
// ---------------------------------------------------------------------------
__global__ __launch_bounds__(256) void fc_kernel(const _Float16* __restrict__ X,
                                                 const _Float16* __restrict__ Wh,
                                                 const float* __restrict__ bias,
                                                 float* __restrict__ out) {
    __shared__ _Float16 Xs[128][40];
    __shared__ _Float16 Ws[64][40];

    const int t    = threadIdx.x;
    const int lane = t & 31, wv = t >> 5;
    const int half = lane >> 4, ln = lane & 15;
    const int cb = blockIdx.x & 15;   // EMB/64
    const int rb = blockIdx.x >> 4;   // 4096/128
    const int row0 = rb * 128, col0 = cb * 64;

    v8f c0 = {}, c1 = {}, c2 = {}, c3 = {};

    for (int kc = 0; kc < EMB; kc += 32) {
        __syncthreads();
        {   // Xs: 128 rows x 32 k
            int row = t >> 1, kb = (t & 1) * 16;
            const _Float16* src = X + (size_t)(row0 + row) * EMB + kc + kb;
#pragma unroll
            for (int j = 0; j < 16; ++j) Xs[row][kb + j] = src[j];
        }
        {   // Ws: 64 cols x 32 k  (B[k][n] = W[n][k], k contiguous)
            int row = t >> 2, kb = (t & 3) * 8;
            const _Float16* src = Wh + (size_t)(col0 + row) * EMB + kc + kb;
#pragma unroll
            for (int j = 0; j < 8; ++j) Ws[row][kb + j] = src[j];
        }
        __syncthreads();

        v16h a = load_a_frag(&Xs[wv * 16][0], 40);
        v16h b;
        b = load_b_frag(&Ws[0][0], 40, 0);   c0 = wmma16x16x32(a, b, c0);
        b = load_b_frag(&Ws[0][0], 40, 16);  c1 = wmma16x16x32(a, b, c1);
        b = load_b_frag(&Ws[0][0], 40, 32);  c2 = wmma16x16x32(a, b, c2);
        b = load_b_frag(&Ws[0][0], 40, 48);  c3 = wmma16x16x32(a, b, c3);
    }

#pragma unroll
    for (int r = 0; r < 8; ++r) {
        const int row = row0 + wv * 16 + r + 8 * half;
        float* op = out + (size_t)row * EMB + col0;
        op[ln]      = c0[r] + bias[col0 + ln];
        op[16 + ln] = c1[r] + bias[col0 + 16 + ln];
        op[32 + ln] = c2[r] + bias[col0 + 32 + ln];
        op[48 + ln] = c3[r] + bias[col0 + 48 + ln];
    }
}

// ---------------------------------------------------------------------------
extern "C" void kernel_launch(void* const* d_in, const int* in_sizes, int n_in,
                              void* d_out, int out_size, void* d_ws, size_t ws_size,
                              hipStream_t stream) {
    const float* value = (const float*)d_in[0];
    const float* key_  = (const float*)d_in[1];
    const float* query = (const float*)d_in[2];
    const int*   mask  = (const int*)d_in[3];
    const float* W_fc  = (const float*)d_in[4];
    const float* b_fc  = (const float*)d_in[5];
    float* out = (float*)d_out;

    const size_t NTOK = (size_t)NB * SEQ;            // 4096
    _Float16* Xh = (_Float16*)d_ws;                  // 4096*1024 f16  (8 MB)
    _Float16* Wh = Xh + NTOK * EMB;                  // 1024*1024 f16  (2 MB)
    _Float16* Kh = Wh + (size_t)EMB * EMB;           // 4096*1024 f16  (8 MB)
    _Float16* Vh = Kh + NTOK * EMB;                  // 4096*1024 f16  (8 MB)
    unsigned* pmask = (unsigned*)(Vh + NTOK * EMB);  // 2*2048*64 u32  (1 MB)

    packmask_kernel<<<(NB * SEQ * SEQ) / 256, 256, 0, stream>>>(mask, pmask);
    convw_kernel<<<(EMB * EMB) / (256 * 4), 256, 0, stream>>>(W_fc, Wh);
    convkv_kernel<<<(int)(NTOK * EMB / (256 * 4)), 256, 0, stream>>>(key_, value, Kh, Vh);
    attn_kernel<<<NB * HEADS * (SEQ / 128), 256, 0, stream>>>(query, Kh, Vh, pmask, Xh);
    fc_kernel<<<(int)(NTOK / 128) * (EMB / 64), 256, 0, stream>>>(Xh, Wh, b_fc, out);
}
